// MultiHeadLatentAttention_43224550867454
// MI455X (gfx1250) — matrix operand order repaired and checked by reading, not compiled, not run on hardware
//
#include <hip/hip_runtime.h>

// ---------------------------------------------------------------------------
// MLA forward for MI455X (gfx1250, wave32, WMMA).
// Pre-pass converts f32 -> f16 (weights transposed to NxK) so every WMMA
// fragment load is a contiguous global_load_b128. GEMM waves compute 32x64
// tiles (8 WMMAs per 12 b128 loads). Attention is flash-style streaming.
// ---------------------------------------------------------------------------

typedef __attribute__((ext_vector_type(16))) _Float16 v16h;
typedef __attribute__((ext_vector_type(8)))  _Float16 v8h;
typedef __attribute__((ext_vector_type(8)))  float    v8f;

#define WMMA_F16(a, b, c) \
    __builtin_amdgcn_wmma_f32_16x16x32_f16(false, (a), false, (b), (short)0, (c), false, false)

namespace mla {
constexpr int Bn  = 2;
constexpr int T   = 2048;
constexpr int D   = 2048;
constexpr int NH  = 16;
constexpr int HD  = 48;
constexpr int LAT = 256;
constexpr int NLH = 8;
constexpr int R   = Bn * T;                    // 4096 rows
constexpr float SCALE = 0.14433756729740643f;  // 1/sqrt(48)
}

// --- fragment load helpers (16B-aligned packed f16) -------------------------
__device__ inline v16h cat8(v8h lo, v8h hi) {
    return __builtin_shufflevector(lo, hi, 0, 1, 2, 3, 4, 5, 6, 7,
                                           8, 9, 10, 11, 12, 13, 14, 15);
}
// 16 contiguous halves (B fragment / V fragment / K fragment)
__device__ inline v16h ld16(const _Float16* p) {
    return cat8(*(const v8h*)p, *(const v8h*)(p + 8));
}
// A fragment: 8 halves at p, 8 halves at p+16 (K-split per §7.12.2)
__device__ inline v16h ldA32(const _Float16* p) {
    return cat8(*(const v8h*)p, *(const v8h*)(p + 16));
}

// ---------------------------------------------------------------------------
// One-time conversion kernels (hoist all f32->f16 cvt out of the GEMM loops).
// ---------------------------------------------------------------------------
__global__ __launch_bounds__(256)
void cvt_f32_to_f16(const float* __restrict__ s, _Float16* __restrict__ d, int n) {
    int i = blockIdx.x * 256 + threadIdx.x;
    if (i < n) d[i] = (_Float16)s[i];
}

// src [K,N] f32 -> dst [N,K] f16 (weight transpose: B fragments become contiguous)
__global__ __launch_bounds__(256)
void cvt_transpose_f32_to_f16(const float* __restrict__ s, _Float16* __restrict__ d,
                              int K, int N) {
    int i = blockIdx.x * 256 + threadIdx.x;
    if (i < N * K) {
        int nrow = i / K;
        int k    = i - nrow * K;
        d[i] = (_Float16)s[(size_t)k * N + nrow];
    }
}

// V [R, NLH*HD] f16 -> Vt [Bn*NLH*HD, T] f16 (PV B-fragment becomes contiguous)
__global__ __launch_bounds__(256)
void transpose_v_f16(const _Float16* __restrict__ V, _Float16* __restrict__ Vt) {
    using namespace mla;
    int i   = blockIdx.x * 256 + threadIdx.x;   // over Bn*NLH*HD*T
    int t   = i & (T - 1);
    int row = i >> 11;                          // i / T
    int b   = row / (NLH * HD);
    int rem = row - b * (NLH * HD);             // lh*HD + d
    Vt[i] = V[(size_t)(b * T + t) * (NLH * HD) + rem];
}

// ---------------------------------------------------------------------------
// GEMM: C[M,N] = alpha * A[M,K] @ B[K,N], A f16 row-major, B given TRANSPOSED
// f16 (Bt[N,K]).  One wave per 32x64 tile: 2x4 WMMA tiles, 8 accumulators,
// per k-step 12 global_load_b128 feed 8 v_wmma (A reused 4x, B reused 2x).
// ---------------------------------------------------------------------------
__global__ __launch_bounds__(32)
void gemm_f16t_wmma(const _Float16* __restrict__ A, const _Float16* __restrict__ Bt,
                    void* __restrict__ Cout, int M, int N, int K,
                    float alpha, int out_f16) {
    const int lane = threadIdx.x & 31;
    const int n    = lane & 15;
    const int h    = lane >> 4;
    const int m0   = blockIdx.y * 32;
    const int n0   = blockIdx.x * 64;

    const _Float16* a0 = A  + (size_t)(m0 + n) * K;        // A row, m = lane&15
    const _Float16* a1 = A  + (size_t)(m0 + 16 + n) * K;
    const _Float16* b0 = Bt + (size_t)(n0 +  0 + n) * K;   // Bt rows, n = lane&15
    const _Float16* b1 = Bt + (size_t)(n0 + 16 + n) * K;
    const _Float16* b2 = Bt + (size_t)(n0 + 32 + n) * K;
    const _Float16* b3 = Bt + (size_t)(n0 + 48 + n) * K;

    v8f acc[2][4] = {};
    for (int k = 0; k < K; k += 32) {
        v16h a[2], b[4];
        a[0] = ld16(a0 + k + h * 16);
        a[1] = ld16(a1 + k + h * 16);
        b[0] = ld16(b0 + k + h * 16);
        b[1] = ld16(b1 + k + h * 16);
        b[2] = ld16(b2 + k + h * 16);
        b[3] = ld16(b3 + k + h * 16);
        if (k + 32 < K) {                       // gfx1250 global_prefetch path
            __builtin_prefetch(a0 + k + 32 + h * 8, 0, 3);
            __builtin_prefetch(b0 + k + 32 + h * 16, 0, 3);
        }
#pragma unroll
        for (int mt = 0; mt < 2; ++mt)
#pragma unroll
            for (int nt = 0; nt < 4; ++nt)
                acc[mt][nt] = WMMA_F16(a[mt], b[nt], acc[mt][nt]);
    }

    if (out_f16) {
        _Float16* C = (_Float16*)Cout;
#pragma unroll
        for (int mt = 0; mt < 2; ++mt)
#pragma unroll
            for (int nt = 0; nt < 4; ++nt)
#pragma unroll
                for (int j = 0; j < 8; ++j)
                    C[(size_t)(m0 + mt * 16 + j + 8 * h) * N + n0 + nt * 16 + n] =
                        (_Float16)(alpha * acc[mt][nt][j]);
    } else {
        float* C = (float*)Cout;
#pragma unroll
        for (int mt = 0; mt < 2; ++mt)
#pragma unroll
            for (int nt = 0; nt < 4; ++nt)
#pragma unroll
                for (int j = 0; j < 8; ++j)
                    C[(size_t)(m0 + mt * 16 + j + 8 * h) * N + n0 + nt * 16 + n] =
                        alpha * acc[mt][nt][j];
    }
}

// ---------------------------------------------------------------------------
// RMS norm (f32 in, f16 out): one 256-thread block per latent row.
// ---------------------------------------------------------------------------
__global__ __launch_bounds__(256)
void rmsnorm256(const float* __restrict__ in, const float* __restrict__ w,
                _Float16* __restrict__ out) {
    const int row = blockIdx.x;
    const int tid = threadIdx.x;
    const float v = in[(size_t)row * mla::LAT + tid];
    __shared__ float red[256];
    red[tid] = v * v;
    __syncthreads();
#pragma unroll
    for (int s = 128; s > 0; s >>= 1) {
        if (tid < s) red[tid] += red[tid + s];
        __syncthreads();
    }
    const float r = rsqrtf(red[0] * (1.0f / mla::LAT) + 1e-5f);
    out[(size_t)row * mla::LAT + tid] = (_Float16)(v * r * w[tid]);
}

// ---------------------------------------------------------------------------
// Flash attention. Grid = (T/16, NH, B); one wave per block.
// Q f16 [R, NH*HD] (pre-scaled by 1/sqrt(HD)), K f16 [R, NLH*HD],
// Vt f16 [Bn*NLH*HD, T], O f16 [R, NH*HD]. Query head h -> latent head h/2.
// ---------------------------------------------------------------------------
__global__ __launch_bounds__(32)
void mla_flash_attn(const _Float16* __restrict__ Q, const _Float16* __restrict__ Kx,
                    const _Float16* __restrict__ Vt, _Float16* __restrict__ O) {
    using namespace mla;
    const int q0   = blockIdx.x * 16;
    const int head = blockIdx.y;
    const int b    = blockIdx.z;
    const int lh   = head >> 1;
    const int lane = threadIdx.x & 31;
    const int nn   = lane & 15;
    const int hf   = lane >> 4;

    __shared__ __align__(16) _Float16 pLds[16 * 32];

    // --- Q fragments: d 0..47, padded to 64 (pad half = zeros) ----------------
    const _Float16* qrow = Q + (size_t)(b * T + q0 + nn) * (NH * HD) + head * HD;
    const v8h z8 = {};
    v16h aq0 = ld16(qrow + hf * 16);                         // d 0..31
    v16h aq1 = hf ? cat8(z8, z8) : ld16(qrow + 32);          // d 32..47 | pad

    v8f o0 = {}, o1 = {}, o2 = {};
    float mrow[8], lrow[8];
#pragma unroll
    for (int j = 0; j < 8; ++j) { mrow[j] = -1e30f; lrow[j] = 0.0f; }

    const _Float16* vhead = Vt + (size_t)(b * NLH + lh) * HD * T;

    const int nkb = (q0 + 16 + 31) / 32;                     // causal: keys [0, q0+16)
    for (int kb = 0; kb < nkb; ++kb) {
        const int c0 = kb * 32;

        // --- scores: two 16x16 tiles, HD padded 48->64 (2 k-steps) ------------
        v8f s0 = {}, s1 = {};
#pragma unroll
        for (int t = 0; t < 2; ++t) {
            const _Float16* kp =
                Kx + (size_t)(b * T + c0 + t * 16 + nn) * (NLH * HD) + lh * HD;
            v16h bk0 = ld16(kp + hf * 16);                   // d 0..31
            v16h bkr = ld16(kp + 32);                        // d 32..47 (in-bounds)
            v16h bk1;
#pragma unroll
            for (int i = 0; i < 16; ++i)                     // hf==1 half is K=48..63 pad
                bk1[i] = hf ? (_Float16)0.0f : bkr[i];
            if (t == 0) { s0 = WMMA_F16(aq0, bk0, s0); s0 = WMMA_F16(aq1, bk1, s0); }
            else        { s1 = WMMA_F16(aq0, bk0, s1); s1 = WMMA_F16(aq1, bk1, s1); }
        }

        // --- causal mask (wave-uniform branch) --------------------------------
        if (c0 + 31 > q0) {
#pragma unroll
            for (int j = 0; j < 8; ++j) {
                int r = q0 + j + 8 * hf;
                if (c0 + nn      > r) s0[j] = -1e9f;
                if (c0 + 16 + nn > r) s1[j] = -1e9f;
            }
        }

        // --- online softmax ----------------------------------------------------
        float p0v[8], p1v[8];
#pragma unroll
        for (int j = 0; j < 8; ++j) {
            float mx = fmaxf(s0[j], s1[j]);
            mx = fmaxf(mx, __shfl_xor(mx, 1, 32));
            mx = fmaxf(mx, __shfl_xor(mx, 2, 32));
            mx = fmaxf(mx, __shfl_xor(mx, 4, 32));
            mx = fmaxf(mx, __shfl_xor(mx, 8, 32));
            float mn = fmaxf(mrow[j], mx);
            float sc = __expf(mrow[j] - mn);
            float p0 = __expf(s0[j] - mn);
            float p1 = __expf(s1[j] - mn);
            float rs = p0 + p1;
            rs += __shfl_xor(rs, 1, 32);
            rs += __shfl_xor(rs, 2, 32);
            rs += __shfl_xor(rs, 4, 32);
            rs += __shfl_xor(rs, 8, 32);
            lrow[j] = lrow[j] * sc + rs;
            mrow[j] = mn;
            o0[j] *= sc; o1[j] *= sc; o2[j] *= sc;
            p0v[j] = p0; p1v[j] = p1;
        }

        // --- P: C-fragment -> LDS -> A-fragment -------------------------------
        __syncthreads();
#pragma unroll
        for (int j = 0; j < 8; ++j) {
            int m = j + 8 * hf;
            pLds[m * 32 + nn]      = (_Float16)p0v[j];
            pLds[m * 32 + 16 + nn] = (_Float16)p1v[j];
        }
        __syncthreads();
        const _Float16* pr = pLds + nn * 32;
        v16h ap = ld16(pr + hf * 16);

        // --- O += P * V (3 n-tiles over HD=48), Vt rows are contiguous in t ---
#pragma unroll
        for (int nt = 0; nt < 3; ++nt) {
            v16h bv = ld16(vhead + (size_t)(nt * 16 + nn) * T + c0 + hf * 16);
            if      (nt == 0) o0 = WMMA_F16(ap, bv, o0);
            else if (nt == 1) o1 = WMMA_F16(ap, bv, o1);
            else              o2 = WMMA_F16(ap, bv, o2);
        }
    }

    // --- epilogue ------------------------------------------------------------
#pragma unroll
    for (int j = 0; j < 8; ++j) {
        const float inv = 1.0f / lrow[j];
        _Float16* orow = O + (size_t)(b * T + q0 + j + 8 * hf) * (NH * HD) + head * HD;
        orow[nn]      = (_Float16)(o0[j] * inv);
        orow[16 + nn] = (_Float16)(o1[j] * inv);
        orow[32 + nn] = (_Float16)(o2[j] * inv);
    }
}

// ---------------------------------------------------------------------------
// Host-side pipeline.
// ---------------------------------------------------------------------------
extern "C" void kernel_launch(void* const* d_in, const int* in_sizes, int n_in,
                              void* d_out, int out_size, void* d_ws, size_t ws_size,
                              hipStream_t stream) {
    (void)in_sizes; (void)n_in; (void)out_size; (void)ws_size;
    using namespace mla;
    const float* x      = (const float*)d_in[0];
    /* d_in[1] = bool causal mask — causality computed analytically, unused */
    const float* Wq     = (const float*)d_in[2];
    const float* Wkv    = (const float*)d_in[3];
    const float* w_norm = (const float*)d_in[4];
    const float* Wk     = (const float*)d_in[5];
    const float* Wv     = (const float*)d_in[6];
    const float* Wo     = (const float*)d_in[7];
    float* out = (float*)d_out;

    // Workspace carve-out (256B aligned), total ~51 MB.
    char* wp = (char*)d_ws;
    auto alloc = [&](size_t bytes) -> char* {
        char* p = wp; wp += (bytes + 255) & ~(size_t)255; return p;
    };
    _Float16* x16   = (_Float16*)alloc((size_t)R * D * 2);          // [4096,2048]
    _Float16* WqT   = (_Float16*)alloc((size_t)(NH * HD) * D * 2);  // [768,2048]
    _Float16* WkvT  = (_Float16*)alloc((size_t)LAT * D * 2);        // [256,2048]
    _Float16* WkT   = (_Float16*)alloc((size_t)(NLH * HD) * LAT * 2);
    _Float16* WvT   = (_Float16*)alloc((size_t)(NLH * HD) * LAT * 2);
    _Float16* WoT   = (_Float16*)alloc((size_t)D * (NH * HD) * 2);  // [2048,768]
    _Float16* Q16   = (_Float16*)alloc((size_t)R * (NH * HD) * 2);
    float*    KVf   = (float*)   alloc((size_t)R * LAT * 4);
    _Float16* KVn16 = (_Float16*)alloc((size_t)R * LAT * 2);
    _Float16* K16   = (_Float16*)alloc((size_t)R * (NLH * HD) * 2);
    _Float16* V16   = (_Float16*)alloc((size_t)R * (NLH * HD) * 2);
    _Float16* Vt16  = (_Float16*)alloc((size_t)Bn * NLH * HD * T * 2);
    _Float16* O16   = (_Float16*)alloc((size_t)R * (NH * HD) * 2);

    auto blk = [](size_t n) { return (unsigned)((n + 255) / 256); };

    // 0) One-time precision/layout conversions
    cvt_f32_to_f16<<<blk((size_t)R * D), 256, 0, stream>>>(x, x16, R * D);
    cvt_transpose_f32_to_f16<<<blk((size_t)(NH * HD) * D), 256, 0, stream>>>(Wq, WqT, D, NH * HD);
    cvt_transpose_f32_to_f16<<<blk((size_t)LAT * D), 256, 0, stream>>>(Wkv, WkvT, D, LAT);
    cvt_transpose_f32_to_f16<<<blk((size_t)(NLH * HD) * LAT), 256, 0, stream>>>(Wk, WkT, LAT, NLH * HD);
    cvt_transpose_f32_to_f16<<<blk((size_t)(NLH * HD) * LAT), 256, 0, stream>>>(Wv, WvT, LAT, NLH * HD);
    cvt_transpose_f32_to_f16<<<blk((size_t)D * (NH * HD)), 256, 0, stream>>>(Wo, WoT, NH * HD, D);

    // 1) Q = (x @ Wq) * 1/sqrt(HD)  -> f16
    gemm_f16t_wmma<<<dim3((NH * HD) / 64, R / 32), 32, 0, stream>>>(
        x16, WqT, Q16, R, NH * HD, D, SCALE, 1);
    // 2) KV = x @ Wkv -> f32 (kept f32 for RMS-norm accuracy)
    gemm_f16t_wmma<<<dim3(LAT / 64, R / 32), 32, 0, stream>>>(
        x16, WkvT, KVf, R, LAT, D, 1.0f, 0);
    // 3) KVn = rms_norm(KV) * w_norm -> f16
    rmsnorm256<<<R, 256, 0, stream>>>(KVf, w_norm, KVn16);
    // 4) K = KVn @ Wk ; V = KVn @ Wv -> f16
    gemm_f16t_wmma<<<dim3((NLH * HD) / 64, R / 32), 32, 0, stream>>>(
        KVn16, WkT, K16, R, NLH * HD, LAT, 1.0f, 1);
    gemm_f16t_wmma<<<dim3((NLH * HD) / 64, R / 32), 32, 0, stream>>>(
        KVn16, WvT, V16, R, NLH * HD, LAT, 1.0f, 1);
    // 5) V transpose for contiguous PV fragments
    transpose_v_f16<<<blk((size_t)Bn * NLH * HD * T), 256, 0, stream>>>(V16, Vt16);
    // 6) Flash attention -> O16
    mla_flash_attn<<<dim3(T / 16, NH, Bn), 32, 0, stream>>>(Q16, K16, Vt16, O16);
    // 7) out = O @ Wo -> f32
    gemm_f16t_wmma<<<dim3(D / 64, R / 32), 32, 0, stream>>>(
        O16, WoT, out, R, D, NH * HD, 1.0f, 0);
}
